// MLSTMFixD_6777458393412
// MI455X (gfx1250) — matrix operand order repaired and checked by reading, not compiled, not run on hardware
//
#include <hip/hip_runtime.h>
#include <cstddef>

// Problem dims (match reference)
#define kT 512
#define kB 64
#define kI 128
#define kH 512
#define kO 128
#define kK 32
#define kNBLK 16          // persistent workgroups (one WGP each)

// LDS ring: [K][B][32+pad] f32, b-stride padded to 34 dwords (bank-conflict free:
// lane group 0-15 hits banks 0-15, group 16-31 (b+8 -> +272 dwords) banks 16-31).
#define RING_BSTRIDE 34
#define RING_SLOT    (kB * RING_BSTRIDE)          // 2176 dwords per slot
#define RING_DWORDS  (kK * RING_SLOT)             // 69632 dwords
#define RING_BYTES   (RING_DWORDS * 4)            // 278528 B  (< 320KB LDS)

typedef __attribute__((ext_vector_type(16))) __bf16        bf16x16;
typedef __attribute__((ext_vector_type(8)))  float         f32x8;
typedef __attribute__((ext_vector_type(4)))  unsigned int  u32x4;

union Frag { u32x4 q[2]; bf16x16 v; };

__device__ __forceinline__ unsigned short f2bf(float f) {
    unsigned int u = __float_as_uint(f);
    u += 0x7FFFu + ((u >> 16) & 1u);     // round-to-nearest-even
    return (unsigned short)(u >> 16);
}
__device__ __forceinline__ float sigmoidf_(float x) {
    return 1.0f / (1.0f + __expf(-x));
}

// Device-scope split barrier over kNBLK resident blocks, one fresh counter per
// step (no sense reversal; counters zeroed at launch).
__device__ __forceinline__ void grid_barrier(int* c) {
    __threadfence();                 // release this thread's global stores
    __syncthreads();
    if (threadIdx.x == 0) {
        __hip_atomic_fetch_add(c, 1, __ATOMIC_RELEASE, __HIP_MEMORY_SCOPE_AGENT);
        while (__hip_atomic_load(c, __ATOMIC_ACQUIRE, __HIP_MEMORY_SCOPE_AGENT) < kNBLK) {
            __builtin_amdgcn_s_sleep(1);
        }
        __threadfence();             // acquire remote stores
    }
    __syncthreads();
}

// ---------------------------------------------------------------- setup kernels

__global__ __launch_bounds__(256) void zero_state_k(float* __restrict__ hf,
                                                    unsigned short* __restrict__ hbf,
                                                    int* __restrict__ ctr,
                                                    int nh, int nc) {
    int i = blockIdx.x * 256 + threadIdx.x;
    if (i < nh) { hf[i] = 0.0f; hbf[i] = 0; }
    if (i < nc) ctr[i] = 0;
}

__global__ __launch_bounds__(256) void cvt_x_k(const float* __restrict__ x,
                                               unsigned short* __restrict__ xb, int n) {
    int i = blockIdx.x * 256 + threadIdx.x;
    if (i < n) xb[i] = f2bf(x[i]);
}

// vtab[L-1][h] = v_L from v_0=1, v_{i+1} = v_i*(i-d)/(i+1), d = 0.5*sigmoid(d_values)
__global__ __launch_bounds__(512) void build_vtab_k(const float* __restrict__ dvals,
                                                    float* __restrict__ vtab) {
    int h = threadIdx.x;              // blockDim = kH = 512
    float dv = 0.5f * sigmoidf_(dvals[h]);
    float v = 1.0f;
    for (int i = 0; i < kK; ++i) {
        v = v * ((float)i - dv) / ((float)(i + 1));
        vtab[i * kH + h] = v;
    }
}

// Pack [W_i | W_o | W_c] (each [640,512] row-major f32) into per-lane bf16 B-fragment
// layout: Wp[ct][kc][lane][dw] (dwords). Column c = ct*16 + lane%16 in [0,1536);
// K row k = kc*32 + (lane>=16)*8 + 2*(dw&3) + (dw>=4)*16 (+0/+1 in low/high bf16).
__global__ __launch_bounds__(256) void pack_gates_k(const float* __restrict__ Wi,
                                                    const float* __restrict__ Wo,
                                                    const float* __restrict__ Wc,
                                                    unsigned int* __restrict__ Wp) {
    int idx = blockIdx.x * 256 + threadIdx.x;
    const int total = 96 * 20 * 32 * 8;
    if (idx >= total) return;
    int dw   = idx & 7;
    int lane = (idx >> 3) & 31;
    int rest = idx >> 8;
    int kc   = rest % 20;
    int ct   = rest / 20;
    int c    = ct * 16 + (lane & 15);
    int gate = c >> 9;
    int hcol = c & 511;
    int k    = kc * 32 + ((lane >> 4) << 3) + (dw & 3) * 2 + ((dw >> 2) << 4);
    const float* W = (gate == 0) ? Wi : ((gate == 1) ? Wo : Wc);
    unsigned short lo = f2bf(W[(size_t)k * kH + hcol]);
    unsigned short hi = f2bf(W[(size_t)(k + 1) * kH + hcol]);
    Wp[idx] = ((unsigned int)hi << 16) | lo;
}

// Pack W_out [512,128] into Wop[nt][kc][lane][dw]
__global__ __launch_bounds__(256) void pack_wout_k(const float* __restrict__ W,
                                                   unsigned int* __restrict__ Wop) {
    int idx = blockIdx.x * 256 + threadIdx.x;
    const int total = 8 * 16 * 32 * 8;
    if (idx >= total) return;
    int dw   = idx & 7;
    int lane = (idx >> 3) & 31;
    int rest = idx >> 8;
    int kc   = rest & 15;
    int nt   = rest >> 4;
    int n    = nt * 16 + (lane & 15);
    int k    = kc * 32 + ((lane >> 4) << 3) + (dw & 3) * 2 + ((dw >> 2) << 4);
    unsigned short lo = f2bf(W[(size_t)k * kO + n]);
    unsigned short hi = f2bf(W[(size_t)(k + 1) * kO + n]);
    Wop[idx] = ((unsigned int)hi << 16) | lo;
}

// ---------------------------------------------------------------- persistent scan

// One 16x16 output tile of out = h @ W_out + b_out for a given wave.
// A and B fragments double-buffered one k-chunk ahead.
__device__ __forceinline__ void out_proj_tile(const unsigned short* __restrict__ h_src,
                                              const unsigned int* __restrict__ Wop,
                                              const float* __restrict__ bout,
                                              float* __restrict__ dst,
                                              int tile, int lane) {
    int omt = tile >> 3, ont = tile & 7;
    int lhi = lane >> 4, ln = lane & 15;
    f32x8 oacc = {};
    const unsigned short* ohrow = h_src + (size_t)(omt * 16 + ln) * kH + lhi * 8;
    const unsigned int* bpo = Wop + (((size_t)(ont * 16) * 32 + lane) << 3);

    Frag A0, A1, B0, B1;
    A0.q[0] = *(const u32x4*)(ohrow);
    A0.q[1] = *(const u32x4*)(ohrow + 16);
    B0.q[0] = *(const u32x4*)(bpo);
    B0.q[1] = *(const u32x4*)(bpo + 4);
#pragma unroll
    for (int kc = 0; kc < 15; ++kc) {
        A1.q[0] = *(const u32x4*)(ohrow + (kc + 1) * 32);
        A1.q[1] = *(const u32x4*)(ohrow + (kc + 1) * 32 + 16);
        const unsigned int* bp = bpo + ((kc + 1) << 8);
        B1.q[0] = *(const u32x4*)bp;
        B1.q[1] = *(const u32x4*)(bp + 4);
        oacc = __builtin_amdgcn_wmma_f32_16x16x32_bf16(
            false, A0.v, false, B0.v, (short)0, oacc, false, false);
        A0 = A1;
        B0 = B1;
    }
    oacc = __builtin_amdgcn_wmma_f32_16x16x32_bf16(
        false, A0.v, false, B0.v, (short)0, oacc, false, false);

    int col  = ont * 16 + ln;
    float vb = bout[col];
#pragma unroll
    for (int r = 0; r < 8; ++r) {
        int b = omt * 16 + lhi * 8 + r;
        dst[(size_t)b * kO + col] = oacc[r] + vb;
    }
}

// Entire T=512 scan in one persistent kernel: 16 blocks x 256 threads (8 waves),
// one block per WGP (forced by 272KB LDS ring). Per step: gate GEMM via
// v_wmma_f32_16x16x32_bf16 (wave w -> M-tile w>>1, h-tile w&1, 3 gate
// accumulators; the 60 t-invariant B fragments get register-cached by the
// compiler via VGPR-MSB addressing), fused out_{t-1} projection on waves 0..1,
// K=32-tap fractional filter against the LDS-resident cell ring, ping-ponged h
// write, then a device-scope barrier.
__global__ __launch_bounds__(256, 1) void scan_all_k(
    const unsigned short* __restrict__ x_bf,   // [T,B,I] bf16
    unsigned short* __restrict__ hbf,          // [2][B*H] bf16 ping-pong
    float* __restrict__ hf,                    // [2][B*H] f32 ping-pong
    const unsigned int* __restrict__ Wp,       // packed gate weights
    const unsigned int* __restrict__ Wop,      // packed W_out
    const float* __restrict__ bi, const float* __restrict__ bo,
    const float* __restrict__ bc, const float* __restrict__ bout,
    const float* __restrict__ vtab,            // [K,H]
    float* __restrict__ out,                   // [T,B,O]
    float* __restrict__ out_h,                 // h_last [B,H]
    float* __restrict__ out_hc,                // hc_last [K,B,H]
    int* __restrict__ ctr) {                   // [T] barrier counters (zeroed)
    extern __shared__ float ring[];            // [K][B][RING_BSTRIDE]

    const int lane = threadIdx.x & 31;
    const int wave = threadIdx.x >> 5;
    const int wg   = blockIdx.x;               // 0..15
    const int mt   = wave >> 1;
    const int ht   = wave & 1;
    const int lhi  = lane >> 4;
    const int ln   = lane & 15;
    const int BH   = kB * kH;

    // zero the LDS ring (pre-history cell states are zero)
    for (int i = threadIdx.x; i < RING_DWORDS; i += 256) ring[i] = 0.0f;
    __syncthreads();

    const int brow = mt * 16 + ln;             // A-matrix M row = batch index
    const int hcol = wg * 32 + ht * 16 + ln;   // this thread's global h column
    const int lc   = ht * 16 + ln;             // local h column in [0,32)
    const float vbi = bi[hcol], vbo = bo[hcol], vbc = bc[hcol];

    // t-invariant preloads: fractional weights + packed-W base pointers
    float vreg[kK];
#pragma unroll
    for (int L = 0; L < kK; ++L) vreg[L] = vtab[L * kH + hcol];
    const unsigned int* bpg[3];
#pragma unroll
    for (int g = 0; g < 3; ++g)
        bpg[g] = Wp + (((size_t)((g * 32 + wg * 2 + ht) * 20) * 32 + lane) << 3);

    const int lbase = (mt * 16 + lhi * 8) * RING_BSTRIDE + lc;   // ring addr base

    for (int t = 0; t < kT; ++t) {
        const unsigned short* h_in = hbf + (size_t)(t & 1) * BH;
        unsigned short* h_out_bf   = hbf + (size_t)((t + 1) & 1) * BH;
        float* h_out_f             = hf  + (size_t)((t + 1) & 1) * BH;

        // ---- gate GEMM: z[64x640] @ [Wi|Wo|Wc] columns of this block
        f32x8 acc[3] = {};
        const unsigned short* xrow = x_bf + ((size_t)t * kB + brow) * kI + lhi * 8;
        const unsigned short* hrow = h_in + (size_t)brow * kH + lhi * 8;

        // x part: kc = 0..3 (K offsets 0..127), fully unrolled; B frags are
        // t-invariant so the compiler keeps them register-resident.
#pragma unroll
        for (int kc = 0; kc < 4; ++kc) {
            Frag A;
            A.q[0] = *(const u32x4*)(xrow + kc * 32);
            A.q[1] = *(const u32x4*)(xrow + kc * 32 + 16);
#pragma unroll
            for (int g = 0; g < 3; ++g) {
                const unsigned int* bp = bpg[g] + (kc << 8);
                Frag Bf;
                Bf.q[0] = *(const u32x4*)bp;
                Bf.q[1] = *(const u32x4*)(bp + 4);
                acc[g] = __builtin_amdgcn_wmma_f32_16x16x32_bf16(
                    false, A.v, false, Bf.v, (short)0, acc[g], false, false);
            }
        }
        // h part: kc = 4..19, A and all 3 B fragments double-buffered one chunk
        // ahead so no WMMA waits on a just-issued load.
        {
            Frag A0, A1, B0[3], B1[3];
            A0.q[0] = *(const u32x4*)(hrow);
            A0.q[1] = *(const u32x4*)(hrow + 16);
#pragma unroll
            for (int g = 0; g < 3; ++g) {
                const unsigned int* bp = bpg[g] + (4 << 8);
                B0[g].q[0] = *(const u32x4*)bp;
                B0[g].q[1] = *(const u32x4*)(bp + 4);
            }
#pragma unroll
            for (int kc = 0; kc < 15; ++kc) {
                A1.q[0] = *(const u32x4*)(hrow + (kc + 1) * 32);
                A1.q[1] = *(const u32x4*)(hrow + (kc + 1) * 32 + 16);
#pragma unroll
                for (int g = 0; g < 3; ++g) {
                    const unsigned int* bp = bpg[g] + ((kc + 5) << 8);
                    B1[g].q[0] = *(const u32x4*)bp;
                    B1[g].q[1] = *(const u32x4*)(bp + 4);
                }
#pragma unroll
                for (int g = 0; g < 3; ++g) {
                    acc[g] = __builtin_amdgcn_wmma_f32_16x16x32_bf16(
                        false, A0.v, false, B0[g].v, (short)0, acc[g], false, false);
                }
                A0 = A1;
#pragma unroll
                for (int g = 0; g < 3; ++g) B0[g] = B1[g];
            }
#pragma unroll
            for (int g = 0; g < 3; ++g) {
                acc[g] = __builtin_amdgcn_wmma_f32_16x16x32_bf16(
                    false, A0.v, false, B0[g].v, (short)0, acc[g], false, false);
            }
        }

        // ---- fused out_{t-1} = h_{t-1} @ W_out + b_out (h_in is h_{t-1})
        if (wave < 2 && t > 0) {
            out_proj_tile(h_in, Wop, bout,
                          out + (size_t)(t - 1) * kB * kO, wg * 2 + wave, lane);
        }

        // ---- epilogue: nonlinearities + K-tap fractional filter vs LDS ring
        const int tslot = t % kK;
        float iv[8], ov[8], gv[8], frac[8];
#pragma unroll
        for (int r = 0; r < 8; ++r) {
            iv[r]   = sigmoidf_(acc[0][r] + vbi);
            ov[r]   = sigmoidf_(acc[1][r] + vbo);
            gv[r]   = tanhf(acc[2][r] + vbc);
            frac[r] = 0.0f;
        }
#pragma unroll 4
        for (int L = 1; L <= kK; ++L) {
            int s    = t - L;
            int slot = ((s % kK) + kK) % kK;        // pre-history slots hold zeros
            const float* rp = ring + slot * RING_SLOT + lbase;
            float vL = vreg[L - 1];
#pragma unroll
            for (int r = 0; r < 8; ++r)
                frac[r] += vL * rp[r * RING_BSTRIDE];
        }
        float* wp_ = ring + tslot * RING_SLOT + lbase;
#pragma unroll
        for (int r = 0; r < 8; ++r) {
            float c  = iv[r] * gv[r] - frac[r];
            wp_[r * RING_BSTRIDE] = c;
            float hn = ov[r] * tanhf(c);
            int b    = mt * 16 + lhi * 8 + r;
            h_out_f[(size_t)b * kH + hcol]  = hn;
            h_out_bf[(size_t)b * kH + hcol] = f2bf(hn);
        }

        grid_barrier(ctr + t);      // h_t now visible device-wide (+LDS sync)
    }

    // ---- tail: final projection out_{T-1} (h_{T-1} lives at parity kT&1)
    if (wave < 2) {
        out_proj_tile(hbf + (size_t)(kT & 1) * BH, Wop, bout,
                      out + (size_t)(kT - 1) * kB * kO, wg * 2 + wave, lane);
    }

    // ---- finalize: h_last (grid-strided) + hc_last from this block's LDS slice
    const int gtid = wg * 256 + threadIdx.x;
    const int nthr = kNBLK * 256;
    const float* hfin = hf + (size_t)(kT & 1) * BH;
    for (int i = gtid; i < BH; i += nthr) out_h[i] = hfin[i];
    // block wg owns h columns [wg*32, wg*32+32) for all (j, b)
    for (int i = threadIdx.x; i < kK * kB * 32; i += 256) {
        int j    = i >> 11;            // / (64*32)
        int rem  = i & 2047;
        int b    = rem >> 5;
        int lc2  = rem & 31;
        int slot = (kT - kK + j) % kK;
        out_hc[(size_t)j * BH + (size_t)b * kH + wg * 32 + lc2] =
            ring[slot * RING_SLOT + b * RING_BSTRIDE + lc2];
    }
}

// ---------------------------------------------------------------- launcher

extern "C" void kernel_launch(void* const* d_in, const int* in_sizes, int n_in,
                              void* d_out, int out_size, void* d_ws, size_t ws_size,
                              hipStream_t stream) {
    (void)in_sizes; (void)n_in; (void)out_size; (void)ws_size;
    const float* x    = (const float*)d_in[0];
    const float* Wi   = (const float*)d_in[1];
    const float* bi   = (const float*)d_in[2];
    const float* Wo   = (const float*)d_in[3];
    const float* bo   = (const float*)d_in[4];
    const float* Wc   = (const float*)d_in[5];
    const float* bc   = (const float*)d_in[6];
    const float* Wout = (const float*)d_in[7];
    const float* bout = (const float*)d_in[8];
    const float* dv   = (const float*)d_in[9];

    char* ws = (char*)d_ws;
    size_t off = 0;
    auto carve = [&](size_t bytes) -> void* {
        void* p = ws + off;
        off = (off + bytes + 255) & ~(size_t)255;
        return p;
    };
    unsigned short* x_bf = (unsigned short*)carve((size_t)kT * kB * kI * 2);
    unsigned int*   Wp   = (unsigned int*)carve((size_t)96 * 20 * 32 * 8 * 4);
    unsigned int*   Wop  = (unsigned int*)carve((size_t)8 * 16 * 32 * 8 * 4);
    float*          vtab = (float*)carve((size_t)kK * kH * 4);
    unsigned short* hbf  = (unsigned short*)carve((size_t)2 * kB * kH * 2);
    float*          hf   = (float*)carve((size_t)2 * kB * kH * 4);
    int*            ctr  = (int*)carve((size_t)kT * 4);

    const int BH = kB * kH;

    zero_state_k<<<(2 * BH + 255) / 256, 256, 0, stream>>>(hf, hbf, ctr, 2 * BH, kT);
    cvt_x_k<<<(kT * kB * kI + 255) / 256, 256, 0, stream>>>(x, x_bf, kT * kB * kI);
    build_vtab_k<<<1, 512, 0, stream>>>(dv, vtab);
    pack_gates_k<<<(96 * 20 * 32 * 8 + 255) / 256, 256, 0, stream>>>(Wi, Wo, Wc, Wp);
    pack_wout_k<<<(8 * 16 * 32 * 8 + 255) / 256, 256, 0, stream>>>(Wout, Wop);

    float* out = (float*)d_out;
    scan_all_k<<<kNBLK, 256, RING_BYTES, stream>>>(
        x_bf, hbf, hf, Wp, Wop, bi, bo, bc, bout, vtab,
        out,
        out + (size_t)kT * kB * kO,
        out + (size_t)kT * kB * kO + BH,
        ctr);
}
